// QImageDFRSystem_37108517437481
// MI455X (gfx1250) — matrix operand order repaired and checked by reading, not compiled, not run on hardware
//
#include <hip/hip_runtime.h>

typedef __attribute__((ext_vector_type(16))) _Float16 v16h;
typedef __attribute__((ext_vector_type(8)))  float    v8f;

#define HDIM  128
#define BM    16
#define NSTEP 49
#define NDEEP 5
#define NREG  4          // layers with register-resident weights (layer 4 in LDS)
#define BTOT  8192
#define WSTR  136        // padded f16 row stride for LDS weight copy

// ---------------- native CDNA5 tanh (v_tanh_f32 confirmed in disasm) ------------
__device__ __forceinline__ float fast_tanh(float x) {
#if __has_builtin(__builtin_amdgcn_tanhf)
  return __builtin_amdgcn_tanhf(x);
#elif __has_builtin(__builtin_amdgcn_tanh_f32)
  return __builtin_amdgcn_tanh_f32(x);
#else
  float e = __builtin_amdgcn_exp2f(x * 2.8853900817779268f);  // 2*log2(e)
  return 1.0f - 2.0f * __builtin_amdgcn_rcpf(e + 1.0f);
#endif
}

// ---------------- fake-quant helper (8-bit uniform, RNE like jnp.round) ----------
__device__ __forceinline__ float fq1(float v, float mn, float mx) {
  float s = (mx - mn) * (1.0f / 255.0f);
  float c = fminf(fmaxf(v, mn), mx);
  return (s > 0.f) ? rintf((c - mn) / s) * s + mn : mn;
}

// ---------------- prep: per-tensor min/max + quantize all weights ----------------
__global__ void prep_quant(const float* __restrict__ p1W, const float* __restrict__ W0,
                           const float* __restrict__ Wd, const float* __restrict__ fc1W,
                           float* __restrict__ p1q, float* __restrict__ W0q,
                           _Float16* __restrict__ Wdq, float* __restrict__ fc1q) {
  __shared__ float smn[256], smx[256];
  const int tid = threadIdx.x;
  auto minmax = [&](const float* src, int n, float& mn, float& mx) {
    float lmn = 3.0e38f, lmx = -3.0e38f;
    for (int i = tid; i < n; i += 256) { float v = src[i]; lmn = fminf(lmn, v); lmx = fmaxf(lmx, v); }
    smn[tid] = lmn; smx[tid] = lmx;
    __syncthreads();
    for (int s = 128; s > 0; s >>= 1) {
      if (tid < s) { smn[tid] = fminf(smn[tid], smn[tid + s]); smx[tid] = fmaxf(smx[tid], smx[tid + s]); }
      __syncthreads();
    }
    mn = smn[0]; mx = smx[0];
    __syncthreads();
  };
  float mn, mx;
  // p1_W [49,196]
  minmax(p1W, 49 * 196, mn, mx);
  for (int i = tid; i < 49 * 196; i += 256) p1q[i] = fq1(p1W[i], mn, mx);
  // W0 [128,1]
  minmax(W0, HDIM, mn, mx);
  for (int i = tid; i < HDIM; i += 256) W0q[i] = fq1(W0[i], mn, mx);
  // Wd [5][128][128], per-slice min/max, stored f16 row-major [n][k]
  for (int d = 0; d < NDEEP; ++d) {
    const float* src = Wd + d * HDIM * HDIM;
    minmax(src, HDIM * HDIM, mn, mx);
    for (int i = tid; i < HDIM * HDIM; i += 256)
      Wdq[d * HDIM * HDIM + i] = (_Float16)fq1(src[i], mn, mx);
  }
  // fc1_W [8,128]
  minmax(fc1W, 8 * HDIM, mn, mx);
  for (int i = tid; i < 8 * HDIM; i += 256) fc1q[i] = fq1(fc1W[i], mn, mx);
}

// ---------------- front: pix = quant(x) @ p1q^T + p1_b  ([B,196]->[B,49]) --------
__global__ void pix_kernel(const float* __restrict__ x, const float* __restrict__ p1q,
                           const float* __restrict__ p1b, float* __restrict__ pix) {
  __shared__ float xs[196];
  const int b = blockIdx.x;
  for (int p = threadIdx.x; p < 196; p += 64) {
    float v = x[b * 196 + p];
    const float s = 4.0f / 255.0f;           // (3.55 - (-0.45)) / 255
    float c = fminf(fmaxf(v, -0.45f), 3.55f);
    xs[p] = rintf((c + 0.45f) / s) * s - 0.45f;
  }
  __syncthreads();
  const int q = threadIdx.x;
  if (q < NSTEP) {
    float acc = p1b[q];
    const float* wrow = p1q + q * 196;
    for (int p = 0; p < 196; ++p) acc += xs[p] * wrow[p];
    pix[b * NSTEP + q] = acc;
  }
}

// ---------------- main: 49-step, 5-layer DFR recurrence + FC head ----------------
// One block = 16 batch rows, 8 waves. Wave w owns output columns [16w, 16w+16).
// Layers 0..3: weights in registers (16 x v16h). Layer 4: weights in LDS
// (padded copy) to keep the wave under 256 VGPRs (no s_set_vgpr_msb, 4 waves/SIMD).
__global__ __launch_bounds__(256, 4) void dfr_kernel(
    const float* __restrict__ pixg, const float* __restrict__ cell,
    const float* __restrict__ W0q, const float* __restrict__ a0p,
    const _Float16* __restrict__ Wdq, const float* __restrict__ adp,
    const float* __restrict__ fc1q, const float* __restrict__ fc2W,
    const float* __restrict__ fc2b, float* __restrict__ out) {
  __shared__ alignas(16) _Float16 wlds[NDEEP - NREG][HDIM][WSTR];  // layer-4 weights
  __shared__ alignas(16) _Float16 bufA[BM][136];   // activation ping
  __shared__ alignas(16) _Float16 bufB[BM][136];   // activation pong
  __shared__ float pixs[BM][52];
  __shared__ float lastb[BM][132];
  __shared__ float out1[BM][8];

  const int tid  = threadIdx.x;
  const int wave = tid >> 5;
  const int lane = tid & 31;
  const int half = lane >> 4;    // A/B fragment half (ISA 16-bit layouts)
  const int ln   = lane & 15;
  const int col  = wave * 16 + ln;      // output column owned by this lane
  const int b0   = blockIdx.x * BM;

  // stage pix block [16,49] into LDS
  for (int i = tid; i < BM * NSTEP; i += 256) {
    int r = i / NSTEP, q = i - r * NSTEP;
    pixs[r][q] = pixg[(b0 + r) * NSTEP + q];
  }

  // stage layer-4 weights into padded LDS (16B chunks)
  for (int i = tid; i < (NDEEP - NREG) * HDIM * (HDIM / 8); i += 256) {
    const int l = i / (HDIM * (HDIM / 8));
    const int rem = i - l * (HDIM * (HDIM / 8));
    const int n = rem / (HDIM / 8);
    const int c = rem - n * (HDIM / 8);
    *(uint4*)&wlds[l][n][c * 8] =
        *(const uint4*)(Wdq + ((NREG + l) * HDIM + n) * HDIM + c * 8);
  }

  // preload register-resident B (weight) fragments for layers 0..3.
  // B fragment (contiguous k, same order as A): lane holds column n=col, K = kt*32 + 16*half + [0..15]
  v16h bw[NREG][4];
#pragma unroll
  for (int d = 0; d < NREG; ++d) {
#pragma unroll
    for (int kt = 0; kt < 4; ++kt) {
      const _Float16* src = Wdq + ((d * HDIM + col) * HDIM + kt * 32 + half * 16);
      union { uint4 u[2]; v16h v; } u;
      u.u[0] = *(const uint4*)(src);
      u.u[1] = *(const uint4*)(src + 8);
      bw[d][kt] = u.v;
    }
  }

  const float w0v = W0q[col];
  const float a0  = a0p[0];
  float ad[NDEEP];
#pragma unroll
  for (int i = 0; i < NDEEP; ++i) ad[i] = adp[i];

  // initial states from cell_out (C/D tile layout: element r -> row r+8*half, col)
  float h0acc[8];
  v8f hs[NDEEP];
#pragma unroll
  for (int r = 0; r < 8; ++r) {
    const int b = b0 + r + 8 * half;
    h0acc[r] = cell[(size_t)b * HDIM + col];
#pragma unroll
    for (int i = 0; i < NDEEP; ++i)
      hs[i][r] = cell[((size_t)(i + 1) * BTOT + b) * HDIM + col];
  }

  __syncthreads();

  for (int t = 0; t < NSTEP; ++t) {
    // shallow cell: h0 = tanh(px * W0q + a0*h0)  (rank-1, pure VALU)
#pragma unroll
    for (int r = 0; r < 8; ++r) {
      const int row = r + 8 * half;
      float v = fast_tanh(pixs[row][t] * w0v + a0 * h0acc[r]);
      h0acc[r] = v;
      bufA[row][col] = (_Float16)v;
    }
    __syncthreads();

    // 5 deep cells: hi = tanh(prev @ Wdq[i]^T + ad[i]*hs[i])
#pragma unroll
    for (int i = 0; i < NDEEP; ++i) {
      const _Float16 (*src)[136] = (i & 1) ? bufB : bufA;
      _Float16 (*dst)[136]       = (i & 1) ? bufA : bufB;
      // A fragment (contiguous k, same order as B): lane holds row ln, K = kt*32 + 16*half + [0..15]
      v16h af[4];
#pragma unroll
      for (int kt = 0; kt < 4; ++kt) {
        const _Float16* p = &src[ln][kt * 32 + 16 * half];
        union { uint4 u[2]; v16h v; } u;
        u.u[0] = *(const uint4*)(p);
        u.u[1] = *(const uint4*)(p + 8);
        af[kt] = u.v;
      }
      v8f c;
#pragma unroll
      for (int r = 0; r < 8; ++r) c[r] = ad[i] * hs[i][r];
      if (i < NREG) {
#pragma unroll
        for (int kt = 0; kt < 4; ++kt)
          c = __builtin_amdgcn_wmma_f32_16x16x32_f16(false, af[kt], false, bw[i][kt],
                                                     (short)0, c, false, false);
      } else {
#pragma unroll
        for (int kt = 0; kt < 4; ++kt) {
          const _Float16* p = &wlds[i - NREG][col][kt * 32 + half * 16];
          union { uint4 u[2]; v16h v; } u;
          u.u[0] = *(const uint4*)(p);
          u.u[1] = *(const uint4*)(p + 8);
          c = __builtin_amdgcn_wmma_f32_16x16x32_f16(false, af[kt], false, u.v,
                                                     (short)0, c, false, false);
        }
      }
#pragma unroll
      for (int r = 0; r < 8; ++r) c[r] = fast_tanh(c[r]);
      hs[i] = c;
      if (i < NDEEP - 1) {
#pragma unroll
        for (int r = 0; r < 8; ++r)
          dst[r + 8 * half][col] = (_Float16)c[r];
      }
      __syncthreads();
    }
  }

  // head: out = relu(last @ fc1q^T) @ fc2^T + fc2_b
#pragma unroll
  for (int r = 0; r < 8; ++r)
    lastb[r + 8 * half][col] = hs[NDEEP - 1][r];
  __syncthreads();

  if (tid < BM * 8) {
    const int row = tid >> 3, o = tid & 7;
    float acc = 0.f;
    const float* w = fc1q + o * HDIM;
    for (int h = 0; h < HDIM; ++h) acc += lastb[row][h] * w[h];
    out1[row][o] = fmaxf(acc, 0.f);
  }
  __syncthreads();
  if (tid < BM * 10) {
    const int row = tid / 10, j = tid - row * 10;
    float acc = fc2b[j];
#pragma unroll
    for (int o = 0; o < 8; ++o) acc += out1[row][o] * fc2W[j * 8 + o];
    out[(size_t)(b0 + row) * 10 + j] = acc;
  }
}

// ---------------- host launcher ----------------
extern "C" void kernel_launch(void* const* d_in, const int* in_sizes, int n_in,
                              void* d_out, int out_size, void* d_ws, size_t ws_size,
                              hipStream_t stream) {
  const float* x    = (const float*)d_in[0];
  const float* cell = (const float*)d_in[1];
  const float* W0   = (const float*)d_in[2];
  const float* a0   = (const float*)d_in[3];
  const float* Wd   = (const float*)d_in[4];
  const float* ad   = (const float*)d_in[5];
  const float* p1W  = (const float*)d_in[6];
  const float* p1b  = (const float*)d_in[7];
  const float* fc1W = (const float*)d_in[8];
  const float* fc2W = (const float*)d_in[9];
  const float* fc2b = (const float*)d_in[10];
  float* out = (float*)d_out;

  char* ws = (char*)d_ws;
  float*    pix  = (float*)(ws);                 // 8192*49 f32  = 1,605,632 B
  float*    p1q  = (float*)(ws + 1605632);       // 49*196 f32   =    38,416 B
  float*    W0q  = (float*)(ws + 1644048);       // 128 f32      =       512 B
  float*    fc1q = (float*)(ws + 1644560);       // 8*128 f32    =     4,096 B
  _Float16* Wdq  = (_Float16*)(ws + 1648656);    // 5*128*128 f16=   163,840 B

  prep_quant<<<1, 256, 0, stream>>>(p1W, W0, Wd, fc1W, p1q, W0q, Wdq, fc1q);
  pix_kernel<<<BTOT, 64, 0, stream>>>(x, p1q, p1b, pix);
  dfr_kernel<<<BTOT / BM, 256, 0, stream>>>(pix, cell, W0q, a0, Wdq, ad,
                                            fc1q, fc2W, fc2b, out);
}
